// SequenceGenerator_7756710937132
// MI455X (gfx1250) — compile-verified, matrix operand order checked by reading, not run on hardware
//
#include <hip/hip_runtime.h>
#include <hip/hip_bf16.h>

#define VOCAB   32000
#define HID     512
#define BATCH   64
#define BEAM    4
#define MAXLEN  32
#define BB      (BATCH * BEAM)   // 256
#define GN      (4 * HID)        // 2048 gate columns
#define TOKLD   (MAXLEN + 1)     // 33
#define NEGF    (-1.0e30f)

typedef __attribute__((ext_vector_type(16))) __bf16 v16bf;
typedef __attribute__((ext_vector_type(8)))  float  v8f;

// ---------------------------------------------------------------------------
// WMMA fragment load (wave32). 16x32 bf16 fragment, row-major source, ld given.
// Lane L: row = L&15 ; elements 0..7 -> K = (L>>4)*8 + 0..7
//                      elements 8..15 -> K = 16 + (L>>4)*8 + 0..7
// Same striping serves A (row = M) and N-major-stored B (row = N).
// ---------------------------------------------------------------------------
__device__ __forceinline__ v16bf load_tile16x32(const __bf16* p0, int ld) {
    int lane = threadIdx.x & 31;
    const __bf16* p = p0 + (long)(lane & 15) * ld + ((lane >> 4) << 3);
    v16bf f;
#pragma unroll
    for (int j = 0; j < 8; ++j) f[j] = p[j];
#pragma unroll
    for (int j = 0; j < 8; ++j) f[j + 8] = p[16 + j];
    return f;
}

// ---------------------------------------------------------------------------
// Async global -> LDS staging (ASYNCcnt).  16 bytes/lane/op.
// ---------------------------------------------------------------------------
__device__ __forceinline__ void stage_async(const __bf16* g, __bf16* s, int bytes) {
    int chunks = bytes >> 4;
    for (int i = threadIdx.x; i < chunks; i += blockDim.x) {
        unsigned lds = (unsigned)(uintptr_t)s + i * 16;
        unsigned long long ga = (unsigned long long)(uintptr_t)g + (unsigned long long)i * 16;
        asm volatile("global_load_async_to_lds_b128 %0, %1, off"
                     :: "v"(lds), "v"(ga) : "memory");
    }
}

__device__ __forceinline__ void wait_async_and_sync() {
    asm volatile("s_wait_asynccnt 0x0" ::: "memory");
    __syncthreads();
}

// Issue async loads of one k-step's 4 B-tiles (16 rows x 32 cols each, 4 KB total)
// into this wave's LDS staging buffer. 8 ops/lane, chunk c -> tile/row/part.
__device__ __forceinline__ void issue_b_tiles(const __bf16* BT, long tileN, int kb,
                                              __bf16* dst /* LDS, 2048 elems */) {
    int lane = threadIdx.x & 31;
#pragma unroll
    for (int rep = 0; rep < 8; ++rep) {
        int c = rep * 32 + lane;       // 0..255 chunks of 16B
        int t = c >> 6;                // tile (64 chunks = 1 KB each)
        int n = (c >> 2) & 15;         // row within tile
        int p = c & 3;                 // 16B part within 64B row
        unsigned long long ga =
            (unsigned long long)(uintptr_t)(BT + (tileN + t * 16 + n) * (long)HID + kb)
            + (unsigned long long)(p * 16);
        unsigned lds = (unsigned)(uintptr_t)dst + (unsigned)(c * 16);
        asm volatile("global_load_async_to_lds_b128 %0, %1, off"
                     :: "v"(lds), "v"(ga) : "memory");
    }
}

// ---------------------------------------------------------------------------
// 16x64 (MxN) x K=HID micro-GEMM, B pipelined through ASYNCcnt double buffer.
//   As : LDS A tile (16 x HID), already staged & synced
//   Bs : per-wave LDS staging, 2 buffers x 2048 elements (8 KB)
// Steady state per k-step: wait asynccnt<=8, 10x ds_load_b128, wait dscnt 0,
// issue 8 async prefetches for k+2, then 4 WMMAs. Global latency spans 2 steps.
// ---------------------------------------------------------------------------
__device__ __forceinline__ void mma_async_16x64(const __bf16* As, __bf16* Bs,
                                                const __bf16* BT, long tileN, v8f acc[4]) {
    issue_b_tiles(BT, tileN, 0, Bs);
    issue_b_tiles(BT, tileN, 32, Bs + 2048);
#pragma unroll
    for (int kb = 0; kb < HID; kb += 32) {
        __bf16* buf = Bs + ((kb >> 5) & 1) * 2048;
        asm volatile("s_wait_asynccnt 0x8" ::: "memory");   // current buffer complete
        v16bf a = load_tile16x32(As + kb, HID);
        v16bf b[4];
#pragma unroll
        for (int t = 0; t < 4; ++t) b[t] = load_tile16x32(buf + t * 512, 32);
        asm volatile("s_wait_dscnt 0x0" ::: "memory");      // reads done before overwrite
        if (kb + 64 < HID) issue_b_tiles(BT, tileN, kb + 64, buf);
#pragma unroll
        for (int t = 0; t < 4; ++t)
            acc[t] = __builtin_amdgcn_wmma_f32_16x16x32_bf16(
                false, a, false, b[t], (short)0, acc[t], false, false);
    }
    asm volatile("s_wait_asynccnt 0x0" ::: "memory");       // drain before reuse/exit
}

// ---------------------------------------------------------------------------
// One-time conversions
// ---------------------------------------------------------------------------
__global__ void transpose_bf16_kernel(const float* __restrict__ in,
                                      __bf16* __restrict__ out, int K, int N) {
    long i = (long)blockIdx.x * blockDim.x + threadIdx.x;
    long total = (long)K * N;
    if (i < total) {
        int n = (int)(i % N);
        int k = (int)(i / N);
        out[(long)n * K + k] = (__bf16)in[i];
    }
}

__global__ void init_kernel(int* tokA, float* prev, float* h, float* c,
                            const float* __restrict__ h0, const float* __restrict__ c0) {
    int i = blockIdx.x * blockDim.x + threadIdx.x;
    if (i < BB * TOKLD) tokA[i] = 1;              // SOS
    if (i < BB) prev[i] = 0.0f;
    if (i < BB * HID) { h[i] = h0[i]; c[i] = c0[i]; }
}

// ---------------------------------------------------------------------------
// Per-step kernels
// ---------------------------------------------------------------------------
__global__ void prep_kernel(const float* __restrict__ emb, const int* __restrict__ tok,
                            int pos, const float* __restrict__ h,
                            __bf16* __restrict__ xb, __bf16* __restrict__ hb) {
    int r = blockIdx.x;
    int t = tok[r * TOKLD + pos];
    const float* erow = emb + (long)t * HID;
    for (int k = threadIdx.x; k < HID; k += blockDim.x) {
        xb[(long)r * HID + k] = (__bf16)erow[k];
        hb[(long)r * HID + k] = (__bf16)h[(long)r * HID + k];
    }
}

// gates = x@Wx + h@Wh + b   (256 x 2048), block = 4 waves, 16x256 tile
__global__ void gates_gemm_kernel(const __bf16* __restrict__ xb, const __bf16* __restrict__ hb,
                                  const __bf16* __restrict__ WxT, const __bf16* __restrict__ WhT,
                                  const float* __restrict__ bias, float* __restrict__ gates) {
    __shared__ __bf16 As[2 * 16 * HID];       // x tile then h tile (32 KB)
    __shared__ __bf16 Bs[4 * 2 * 2048];       // per-wave double buffers (32 KB)
    int wave  = threadIdx.x >> 5;
    long tileM = (long)blockIdx.x * 16;
    long tileN = (long)blockIdx.y * 256 + wave * 64;
    __bf16* myB = Bs + wave * 4096;

    stage_async(xb + tileM * HID, As, 16 * HID * (int)sizeof(__bf16));
    stage_async(hb + tileM * HID, As + 16 * HID, 16 * HID * (int)sizeof(__bf16));
    wait_async_and_sync();

    v8f acc[4] = {v8f{}, v8f{}, v8f{}, v8f{}};
    mma_async_16x64(As, myB, WxT, tileN, acc);
    mma_async_16x64(As + 16 * HID, myB, WhT, tileN, acc);

    int lane = threadIdx.x & 31;
    int n  = lane & 15;
    int mb = (lane >> 4) << 3;
#pragma unroll
    for (int t = 0; t < 4; ++t) {
        long ncol = tileN + t * 16 + n;
        float bv = bias[ncol];
#pragma unroll
        for (int j = 0; j < 8; ++j)
            gates[(tileM + mb + j) * GN + ncol] = acc[t][j] + bv;
    }
}

__device__ __forceinline__ float sigm(float x) { return 1.0f / (1.0f + __expf(-x)); }

__global__ void lstm_kernel(const float* __restrict__ gates, const float* __restrict__ c,
                            float* __restrict__ h2, float* __restrict__ c2,
                            __bf16* __restrict__ h2b) {
    int r = blockIdx.x;
    for (int t = threadIdx.x; t < HID; t += blockDim.x) {
        long g0 = (long)r * GN + t;
        float gi = gates[g0];
        float gf = gates[g0 + HID];
        float gg = gates[g0 + 2 * HID];
        float go = gates[g0 + 3 * HID];
        float cn = sigm(gf) * c[(long)r * HID + t] + sigm(gi) * tanhf(gg);
        float hn = sigm(go) * tanhf(cn);
        c2[(long)r * HID + t] = cn;
        h2[(long)r * HID + t] = hn;
        h2b[(long)r * HID + t] = (__bf16)hn;
    }
}

// logits = h2 @ Wout + bout  (256 x 32000), PAD/SOS/EOS masked to NEG
__global__ void logits_gemm_kernel(const __bf16* __restrict__ h2b,
                                   const __bf16* __restrict__ WoutT,
                                   const float* __restrict__ bout,
                                   float* __restrict__ logits) {
    __shared__ __bf16 As[16 * HID];           // 16 KB
    __shared__ __bf16 Bs[4 * 2 * 2048];       // 32 KB
    int wave  = threadIdx.x >> 5;
    long tileM = (long)blockIdx.x * 16;
    long tileN = (long)blockIdx.y * 256 + wave * 64;
    __bf16* myB = Bs + wave * 4096;

    stage_async(h2b + tileM * HID, As, 16 * HID * (int)sizeof(__bf16));
    wait_async_and_sync();

    v8f acc[4] = {v8f{}, v8f{}, v8f{}, v8f{}};
    mma_async_16x64(As, myB, WoutT, tileN, acc);

    int lane = threadIdx.x & 31;
    int n  = lane & 15;
    int mb = (lane >> 4) << 3;
#pragma unroll
    for (int t = 0; t < 4; ++t) {
        long ncol = tileN + t * 16 + n;
        float bv = bout[ncol];
        bool masked = (ncol < 3);
#pragma unroll
        for (int j = 0; j < 8; ++j) {
            float v = acc[t][j] + bv;
            logits[(tileM + mb + j) * VOCAB + ncol] = masked ? NEGF : v;
        }
    }
}

// per-row logsumexp over VOCAB
__global__ void lse_kernel(const float* __restrict__ logits, float* __restrict__ row_lse) {
    int r = blockIdx.x;
    const float* row = logits + (long)r * VOCAB;
    __shared__ float red[256];
    float m = NEGF;
    for (int v = threadIdx.x; v < VOCAB; v += 256) m = fmaxf(m, row[v]);
    red[threadIdx.x] = m; __syncthreads();
    for (int s = 128; s > 0; s >>= 1) {
        if (threadIdx.x < s) red[threadIdx.x] = fmaxf(red[threadIdx.x], red[threadIdx.x + s]);
        __syncthreads();
    }
    m = red[0]; __syncthreads();
    float sum = 0.0f;
    for (int v = threadIdx.x; v < VOCAB; v += 256) sum += __expf(row[v] - m);
    red[threadIdx.x] = sum; __syncthreads();
    for (int s = 128; s > 0; s >>= 1) {
        if (threadIdx.x < s) red[threadIdx.x] += red[threadIdx.x + s];
        __syncthreads();
    }
    if (threadIdx.x == 0) row_lse[r] = m + __logf(red[0]);
}

// per-batch top-4 over BEAM*VOCAB candidates; tree-merge of sorted top-4 lists
__global__ void topk_kernel(const float* __restrict__ logits, const float* __restrict__ row_lse,
                            const float* __restrict__ prev, int pos,
                            float* __restrict__ top_s, int* __restrict__ top_i) {
    int b = blockIdx.x, tid = threadIdx.x;   // 256 threads
    float s[4] = {NEGF * 2.0f, NEGF * 2.0f, NEGF * 2.0f, NEGF * 2.0f};
    int   id[4] = {0, 0, 0, 0};
    for (int beam = 0; beam < BEAM; ++beam) {
        int r = b * BEAM + beam;
        float add = (pos == 0) ? (beam == 0 ? 0.0f : NEGF) : prev[r];
        float lse = row_lse[r];
        const float* lrow = logits + (long)r * VOCAB;
        for (int v = tid; v < VOCAB; v += 256) {
            float val = lrow[v] - lse + add;
            if (val > s[3]) {
                s[3] = val; id[3] = beam * VOCAB + v;
                for (int q = 3; q > 0 && s[q] > s[q - 1]; --q) {
                    float ts = s[q]; s[q] = s[q - 1]; s[q - 1] = ts;
                    int ti = id[q]; id[q] = id[q - 1]; id[q - 1] = ti;
                }
            }
        }
    }
    __shared__ float ss[1024];
    __shared__ int   si[1024];
#pragma unroll
    for (int j = 0; j < 4; ++j) { ss[tid * 4 + j] = s[j]; si[tid * 4 + j] = id[j]; }
    __syncthreads();
    for (int step = 128; step >= 1; step >>= 1) {
        if (tid < step) {
            float ls[4]; int li[4];
#pragma unroll
            for (int j = 0; j < 4; ++j) { ls[j] = ss[tid * 4 + j]; li[j] = si[tid * 4 + j]; }
#pragma unroll
            for (int j = 0; j < 4; ++j) {
                float v = ss[(tid + step) * 4 + j];
                int  vi = si[(tid + step) * 4 + j];
                if (v > ls[3]) {
                    ls[3] = v; li[3] = vi;
                    for (int q = 3; q > 0 && ls[q] > ls[q - 1]; --q) {
                        float ts = ls[q]; ls[q] = ls[q - 1]; ls[q - 1] = ts;
                        int ti = li[q]; li[q] = li[q - 1]; li[q - 1] = ti;
                    }
                }
            }
#pragma unroll
            for (int j = 0; j < 4; ++j) { ss[tid * 4 + j] = ls[j]; si[tid * 4 + j] = li[j]; }
        }
        __syncthreads();
    }
    if (tid < 4) {
        top_s[b * 4 + tid] = ss[tid];
        top_i[b * 4 + tid] = si[tid];
    }
}

// gather beams: tokens, h, c, scores
__global__ void update_kernel(int pos, const int* __restrict__ tin, int* __restrict__ tout,
                              const float* __restrict__ h2, const float* __restrict__ c2,
                              float* __restrict__ h, float* __restrict__ c,
                              const float* __restrict__ top_s, const int* __restrict__ top_i,
                              float* __restrict__ prev) {
    int r = blockIdx.x;
    int b = r >> 2, j = r & 3;
    int idx  = top_i[b * 4 + j];
    int beam = idx / VOCAB;
    int word = idx - beam * VOCAB;
    int cand = b * 4 + beam;
    for (int t = threadIdx.x; t < TOKLD; t += blockDim.x)
        tout[r * TOKLD + t] = tin[cand * TOKLD + t];
    for (int t = threadIdx.x; t < HID; t += blockDim.x) {
        h[(long)r * HID + t] = h2[(long)cand * HID + t];
        c[(long)r * HID + t] = c2[(long)cand * HID + t];
    }
    __syncthreads();
    if (threadIdx.x == 0) {
        tout[r * TOKLD + pos + 1] = word;
        prev[r] = top_s[b * 4 + j];
    }
}

__global__ void final_kernel(const int* __restrict__ tok, const float* __restrict__ prev,
                             int* __restrict__ out_tok, float* __restrict__ out_sc) {
    int b = blockIdx.x;  // 64 blocks, 32 threads
    __shared__ int bestj;
    if (threadIdx.x == 0) {
        float best = NEGF * 4.0f; int bj = 0;
        for (int j = 0; j < BEAM; ++j) {
            float v = prev[b * BEAM + j];
            if (v > best) { best = v; bj = j; }
        }
        bestj = bj;
        out_sc[b] = best;
    }
    __syncthreads();
    int r = b * BEAM + bestj;
    for (int t = threadIdx.x; t < MAXLEN; t += blockDim.x)
        out_tok[b * MAXLEN + t] = tok[r * TOKLD + 1 + t];
}

// ---------------------------------------------------------------------------
extern "C" void kernel_launch(void* const* d_in, const int* in_sizes, int n_in,
                              void* d_out, int out_size, void* d_ws, size_t ws_size,
                              hipStream_t stream) {
    (void)in_sizes; (void)n_in; (void)out_size; (void)ws_size;
    const float* emb  = (const float*)d_in[0];
    const float* Wx   = (const float*)d_in[1];
    const float* Wh   = (const float*)d_in[2];
    const float* bias = (const float*)d_in[3];
    const float* Wout = (const float*)d_in[4];
    const float* bout = (const float*)d_in[5];
    const float* h0   = (const float*)d_in[6];
    const float* c0   = (const float*)d_in[7];

    char* ws = (char*)d_ws;
    size_t off = 0;
    auto alloc = [&](size_t bytes) -> void* {
        void* p = ws + off;
        off = (off + bytes + 255) & ~(size_t)255;
        return p;
    };
    __bf16* WoutT  = (__bf16*)alloc((size_t)VOCAB * HID * sizeof(__bf16));
    __bf16* WxT    = (__bf16*)alloc((size_t)GN * HID * sizeof(__bf16));
    __bf16* WhT    = (__bf16*)alloc((size_t)GN * HID * sizeof(__bf16));
    float*  logits = (float*)alloc((size_t)BB * VOCAB * sizeof(float));
    float*  gates  = (float*)alloc((size_t)BB * GN * sizeof(float));
    __bf16* xb     = (__bf16*)alloc((size_t)BB * HID * sizeof(__bf16));
    __bf16* hb     = (__bf16*)alloc((size_t)BB * HID * sizeof(__bf16));
    __bf16* h2b    = (__bf16*)alloc((size_t)BB * HID * sizeof(__bf16));
    float*  hbuf   = (float*)alloc((size_t)BB * HID * sizeof(float));
    float*  cbuf   = (float*)alloc((size_t)BB * HID * sizeof(float));
    float*  h2buf  = (float*)alloc((size_t)BB * HID * sizeof(float));
    float*  c2buf  = (float*)alloc((size_t)BB * HID * sizeof(float));
    int*    tokA   = (int*)alloc((size_t)BB * TOKLD * sizeof(int));
    int*    tokB   = (int*)alloc((size_t)BB * TOKLD * sizeof(int));
    float*  prev   = (float*)alloc((size_t)BB * sizeof(float));
    float*  rowlse = (float*)alloc((size_t)BB * sizeof(float));
    float*  top_s  = (float*)alloc((size_t)BATCH * BEAM * sizeof(float));
    int*    top_i  = (int*)alloc((size_t)BATCH * BEAM * sizeof(int));

    {
        long tot = (long)HID * GN;
        transpose_bf16_kernel<<<(tot + 255) / 256, 256, 0, stream>>>(Wx, WxT, HID, GN);
        transpose_bf16_kernel<<<(tot + 255) / 256, 256, 0, stream>>>(Wh, WhT, HID, GN);
        long totO = (long)HID * VOCAB;
        transpose_bf16_kernel<<<(totO + 255) / 256, 256, 0, stream>>>(Wout, WoutT, HID, VOCAB);
        init_kernel<<<(BB * HID + 255) / 256, 256, 0, stream>>>(tokA, prev, hbuf, cbuf, h0, c0);
    }

    for (int pos = 0; pos < MAXLEN; ++pos) {
        int* tin  = (pos & 1) ? tokB : tokA;
        int* tout = (pos & 1) ? tokA : tokB;
        prep_kernel<<<BB, 256, 0, stream>>>(emb, tin, pos, hbuf, xb, hb);
        gates_gemm_kernel<<<dim3(BB / 16, GN / 256), 128, 0, stream>>>(xb, hb, WxT, WhT, bias, gates);
        lstm_kernel<<<BB, 256, 0, stream>>>(gates, cbuf, h2buf, c2buf, h2b);
        logits_gemm_kernel<<<dim3(BB / 16, VOCAB / 256), 128, 0, stream>>>(h2b, WoutT, bout, logits);
        lse_kernel<<<BB, 256, 0, stream>>>(logits, rowlse);
        topk_kernel<<<BATCH, 256, 0, stream>>>(logits, rowlse, prev, pos, top_s, top_i);
        update_kernel<<<BB, 128, 0, stream>>>(pos, tin, tout, h2buf, c2buf, hbuf, cbuf,
                                              top_s, top_i, prev);
    }

    int* finalTok = ((MAXLEN - 1) & 1) ? tokA : tokB;  // last write target
    final_kernel<<<BATCH, 32, 0, stream>>>(finalTok, prev,
                                           (int*)d_out, (float*)d_out + BATCH * MAXLEN);
}
